// HybridTransformerQuantum_65481071404699
// MI455X (gfx1250) — matrix operand order repaired and checked by reading, not compiled, not run on hardware
//
#include <hip/hip_runtime.h>
#include <hip/hip_bf16.h>

// ---------------------------------------------------------------------------
// HybridTransformerQuantum forward for MI455X (gfx1250, wave32, WMMA f16)
// ---------------------------------------------------------------------------

typedef __attribute__((ext_vector_type(16))) _Float16 v16h;
typedef __attribute__((ext_vector_type(8)))  _Float16 v8h;
typedef __attribute__((ext_vector_type(8)))  float    v8f;

union V16H { v16h v; v8h h2[2]; _Float16 e[16]; };

#define E_DIM 128
#define S_LEN 1024
#define F_DIM 512
#define L_NUM 4
#define BS    32768   // B*S

__device__ __forceinline__ v8f wmma32(v16h a, v16h b, v8f c) {
    return __builtin_amdgcn_wmma_f32_16x16x32_f16(
        /*neg_a=*/false, a, /*neg_b=*/false, b,
        /*c_mod=*/(short)0, c, /*reuse_a=*/false, /*reuse_b=*/false);
}

// A-matrix fragment, 16x32 f16, element (m,k) at p[(row0+m)*ld + k0 + k].
// Per ISA: lanes 0-15 m=0..15; VGPR v<4 -> K=2v+8*half, v>=4 -> K=16+2(v-4)+8*half.
__device__ __forceinline__ v16h load_afrag(const _Float16* __restrict__ p,
                                           int ld, int row0, int k0) {
    int lane = threadIdx.x & 31;
    int half = lane >> 4;
    const _Float16* base = p + (size_t)(row0 + (lane & 15)) * ld + k0 + 8 * half;
    V16H r;
    r.h2[0] = *(const v8h*)(base);        // K = k0+8h .. +7
    r.h2[1] = *(const v8h*)(base + 16);   // K = k0+16+8h .. +7
    return r.v;
}

// B-matrix fragment, 32x16 f16, element (k,n) at p[(col0+n)*ld + k0 + k].
// Same K order as A: lanes n=lane%16, elems 0-7 K=8h..8h+7, elems 8-15 K=16+8h..+7.
__device__ __forceinline__ v16h load_bfrag(const _Float16* __restrict__ p,
                                           int ld, int col0, int k0) {
    int lane = threadIdx.x & 31;
    int half = lane >> 4;
    const _Float16* base = p + (size_t)(col0 + (lane & 15)) * ld + k0 + 8 * half;
    V16H r;
    r.h2[0] = *(const v8h*)(base);
    r.h2[1] = *(const v8h*)(base + 16);
    return r.v;
}

// ---------------------------------------------------------------------------
// 0) One-shot f32 -> f16 weight conversion (weights total ~1.2 MB; converting
//    once removes 16 VALU cvts per B-fragment per K-step from every GEMM tile)
// ---------------------------------------------------------------------------
__global__ __launch_bounds__(256) void cvt_f16_kernel(const float* __restrict__ src,
                                                      _Float16* __restrict__ dst,
                                                      int n) {
    for (int i = blockIdx.x * 256 + threadIdx.x; i < n; i += gridDim.x * 256)
        dst[i] = (_Float16)src[i];
}

// ---------------------------------------------------------------------------
// 1) Embedding + sinusoidal positional encoding -> x (f32 [B*S, E])
// ---------------------------------------------------------------------------
__global__ __launch_bounds__(128) void embed_kernel(const int* __restrict__ tokens,
                                                    const float* __restrict__ emb,
                                                    float* __restrict__ x) {
    int tok_idx = blockIdx.x;            // 0..BS-1
    int e = threadIdx.x;                 // 0..127
    int s = tok_idx & (S_LEN - 1);
    int t = tokens[tok_idx];
    int i2 = e & ~1;
    // div = exp(-(2i)*ln(10000)/E)
    float div = __expf(-(float)i2 * (9.210340371976184f / (float)E_DIM));
    float ang = (float)s * div;
    float pe = (e & 1) ? __cosf(ang) : __sinf(ang);
    x[(size_t)tok_idx * E_DIM + e] = emb[(size_t)t * E_DIM + e] + pe;
}

// ---------------------------------------------------------------------------
// Wave-per-token layernorm helper (wave32)
// ---------------------------------------------------------------------------
__device__ __forceinline__ float wave_sum(float v) {
#pragma unroll
    for (int m = 16; m >= 1; m >>= 1) v += __shfl_xor(v, m, 32);
    return v;
}

// 2) LN1 + cos(h + tile(phi)) -> q [BS,E] f16 and qT per-batch [E,S] f16
__global__ __launch_bounds__(256) void qgen_attn_kernel(const float* __restrict__ x,
                                                        const float* __restrict__ g,
                                                        const float* __restrict__ bt,
                                                        const float* __restrict__ phi,
                                                        _Float16* __restrict__ q,
                                                        _Float16* __restrict__ qT) {
    int wave = threadIdx.x >> 5, lane = threadIdx.x & 31;
    int tok = blockIdx.x * 8 + wave;                 // 0..BS-1
    int b = tok >> 10, s = tok & (S_LEN - 1);
    const float* xr = x + (size_t)tok * E_DIM;
    float v[4], sum = 0.f;
#pragma unroll
    for (int i = 0; i < 4; ++i) { v[i] = xr[lane + 32 * i]; sum += v[i]; }
    float mu = wave_sum(sum) * (1.0f / E_DIM);
    float var = 0.f;
#pragma unroll
    for (int i = 0; i < 4; ++i) { float d = v[i] - mu; var += d * d; }
    var = wave_sum(var) * (1.0f / E_DIM);
    float rstd = rsqrtf(var + 1e-5f);
#pragma unroll
    for (int i = 0; i < 4; ++i) {
        int e = lane + 32 * i;
        float h = (v[i] - mu) * rstd * g[e] + bt[e];
        float qv = __cosf(h + phi[e & 7]);           // tile(phi, H): index e % DK
        q[(size_t)tok * E_DIM + e] = (_Float16)qv;
        qT[((size_t)b * E_DIM + e) * S_LEN + s] = (_Float16)qv;
    }
}

// 5) LN2 + cos(h)*cos(psi) -> qf [BS,E] f16
__global__ __launch_bounds__(256) void qgen_ffn_kernel(const float* __restrict__ x,
                                                       const float* __restrict__ g,
                                                       const float* __restrict__ bt,
                                                       const float* __restrict__ psi,
                                                       _Float16* __restrict__ qf) {
    int wave = threadIdx.x >> 5, lane = threadIdx.x & 31;
    int tok = blockIdx.x * 8 + wave;
    const float* xr = x + (size_t)tok * E_DIM;
    float v[4], sum = 0.f;
#pragma unroll
    for (int i = 0; i < 4; ++i) { v[i] = xr[lane + 32 * i]; sum += v[i]; }
    float mu = wave_sum(sum) * (1.0f / E_DIM);
    float var = 0.f;
#pragma unroll
    for (int i = 0; i < 4; ++i) { float d = v[i] - mu; var += d * d; }
    var = wave_sum(var) * (1.0f / E_DIM);
    float rstd = rsqrtf(var + 1e-5f);
#pragma unroll
    for (int i = 0; i < 4; ++i) {
        int e = lane + 32 * i;
        float h = (v[i] - mu) * rstd * g[e] + bt[e];
        qf[(size_t)tok * E_DIM + e] = (_Float16)(__cosf(h) * __cosf(psi[e]));
    }
}

// ---------------------------------------------------------------------------
// 3) Flash attention. One wave owns a 16-row tile; streams 32-col K-blocks.
//    q==k==v. Scores via 4 chained WMMA (K=E=128); PV via 8 WMMA (N=E/16).
// ---------------------------------------------------------------------------
__global__ __launch_bounds__(256) void attn_kernel(const _Float16* __restrict__ q,
                                                   const _Float16* __restrict__ qT,
                                                   _Float16* __restrict__ out) {
    __shared__ __align__(16) _Float16 plds[8][16 * 32];   // per-wave 16x32 P tile
    int wave = threadIdx.x >> 5, lane = threadIdx.x & 31;
    int half = lane >> 4, l16 = lane & 15;
    int tile = blockIdx.x * 8 + wave;                     // 0..2047
    int b = tile >> 6;                                    // 64 row-tiles per batch
    int m0 = (tile & 63) << 4;
    const _Float16* qb  = q  + (size_t)b * S_LEN * E_DIM;
    const _Float16* qTb = qT + (size_t)b * E_DIM * S_LEN;

    v16h afr[4];
#pragma unroll
    for (int i = 0; i < 4; ++i) afr[i] = load_afrag(qb, E_DIM, m0, i * 32);

    v8f acc[8];
#pragma unroll
    for (int i = 0; i < 8; ++i)
#pragma unroll
        for (int r = 0; r < 8; ++r) acc[i][r] = 0.f;

    float mrun[8], lrun[8];
#pragma unroll
    for (int r = 0; r < 8; ++r) { mrun[r] = -1e30f; lrun[r] = 0.f; }

    const float scale = 0.3535533905932738f;              // 1/sqrt(DK), DK=8

    for (int t0 = 0; t0 < S_LEN; t0 += 32) {
        // --- scores: two 16x16 tiles over columns [t0, t0+32) ---
        v8f s0, s1;
#pragma unroll
        for (int r = 0; r < 8; ++r) { s0[r] = 0.f; s1[r] = 0.f; }
#pragma unroll
        for (int kk = 0; kk < 4; ++kk) {
            v16h b0 = load_bfrag(qb, E_DIM, t0,      kk * 32);
            v16h b1 = load_bfrag(qb, E_DIM, t0 + 16, kk * 32);
            s0 = wmma32(afr[kk], b0, s0);
            s1 = wmma32(afr[kk], b1, s1);
        }
        // --- online softmax update (rows live in 16-lane half-groups) ---
        float p0[8], p1[8], alph[8];
#pragma unroll
        for (int r = 0; r < 8; ++r) {
            float a = s0[r] * scale, c = s1[r] * scale;
            float rm = fmaxf(a, c);
#pragma unroll
            for (int m = 8; m >= 1; m >>= 1) rm = fmaxf(rm, __shfl_xor(rm, m, 32));
            float mn = fmaxf(mrun[r], rm);
            alph[r] = __expf(mrun[r] - mn);
            p0[r] = __expf(a - mn);
            p1[r] = __expf(c - mn);
            float rs = p0[r] + p1[r];
#pragma unroll
            for (int m = 8; m >= 1; m >>= 1) rs += __shfl_xor(rs, m, 32);
            lrun[r] = lrun[r] * alph[r] + rs;
            mrun[r] = mn;
        }
#pragma unroll
        for (int i = 0; i < 8; ++i)
#pragma unroll
            for (int r = 0; r < 8; ++r) acc[i][r] *= alph[r];

        // --- D-layout P -> A-layout via LDS round trip ---
        _Float16* pl = plds[wave];
#pragma unroll
        for (int r = 0; r < 8; ++r) {
            int row = 8 * half + r;
            pl[row * 32 + l16]      = (_Float16)p0[r];
            pl[row * 32 + 16 + l16] = (_Float16)p1[r];
        }
        __syncthreads();
        v16h pfr = load_afrag(pl, 32, 0, 0);
        __syncthreads();

        // --- acc += P(16x32) @ V(32x16 per E-tile), V from qT [E,S] ---
#pragma unroll
        for (int i = 0; i < 8; ++i) {
            v16h vfr = load_bfrag(qTb, S_LEN, i * 16, t0);
            acc[i] = wmma32(pfr, vfr, acc[i]);
        }
    }

    // --- normalize and store f16 ---
    float rinv[8];
#pragma unroll
    for (int r = 0; r < 8; ++r) rinv[r] = 1.0f / lrun[r];
    _Float16* ob = out + (size_t)b * S_LEN * E_DIM;
#pragma unroll
    for (int i = 0; i < 8; ++i)
#pragma unroll
        for (int r = 0; r < 8; ++r) {
            int row = m0 + 8 * half + r;
            ob[(size_t)row * E_DIM + i * 16 + l16] = (_Float16)(acc[i][r] * rinv[r]);
        }
}

// ---------------------------------------------------------------------------
// Generic WMMA GEMM: D[M,N] = act( A[M,K]_f16 @ W[N,K]^T ) (+residual)
// Each wave computes a 16x64 output strip (4 N-tiles): the A-fragment is
// reused by 4 WMMAs per K-step, quadrupling matrix-op density per load.
// All operands are L2-resident at these sizes (192 MB L2).
// ---------------------------------------------------------------------------
template <int N, int K, bool RELU, bool RES, bool OUT16>
__global__ __launch_bounds__(256) void gemm_kernel(const _Float16* __restrict__ A,
                                                   const _Float16* __restrict__ W,
                                                   const float* __restrict__ resid,
                                                   float* __restrict__ outf,
                                                   _Float16* __restrict__ outh) {
    int wave = threadIdx.x >> 5, lane = threadIdx.x & 31;
    int half = lane >> 4, l16 = lane & 15;
    int tile = blockIdx.x * 8 + wave;
    constexpr int TN = N / 64;                 // strips of 4 N-tiles
    int tm = tile / TN, tn = tile % TN;

    v8f acc[4];
#pragma unroll
    for (int j = 0; j < 4; ++j)
#pragma unroll
        for (int r = 0; r < 8; ++r) acc[j][r] = 0.f;

#pragma unroll 4
    for (int k0 = 0; k0 < K; k0 += 32) {
        v16h a = load_afrag(A, K, tm * 16, k0);
#pragma unroll
        for (int j = 0; j < 4; ++j) {
            v16h b = load_bfrag(W, K, tn * 64 + j * 16, k0);
            acc[j] = wmma32(a, b, acc[j]);
        }
    }
#pragma unroll
    for (int j = 0; j < 4; ++j)
#pragma unroll
        for (int r = 0; r < 8; ++r) {
            int row = tm * 16 + 8 * half + r;
            int col = tn * 64 + j * 16 + l16;
            float v = acc[j][r];
            if (RES) v += resid[(size_t)row * N + col];
            if (RELU) v = fmaxf(v, 0.f);
            if (OUT16) outh[(size_t)row * N + col] = (_Float16)v;
            else       outf[(size_t)row * N + col] = v;
        }
}

// ---------------------------------------------------------------------------
// Final pooling + classifier (tiny)
// ---------------------------------------------------------------------------
__global__ __launch_bounds__(128) void pool_kernel(const float* __restrict__ x,
                                                   float* __restrict__ pooled) {
    int b = blockIdx.x, e = threadIdx.x;
    const float* xb = x + (size_t)b * S_LEN * E_DIM;
    float s = 0.f;
    for (int t = 0; t < S_LEN; ++t) s += xb[(size_t)t * E_DIM + e];
    pooled[b * E_DIM + e] = s * (1.0f / S_LEN);
}

__global__ __launch_bounds__(256) void cls_kernel(const float* __restrict__ pooled,
                                                  const float* __restrict__ Wc,
                                                  const float* __restrict__ bc,
                                                  float* __restrict__ out) {
    int id = threadIdx.x;            // 256 = 32 batches * 8 classes
    int b = id >> 3, c = id & 7;
    float s = bc[c];
#pragma unroll 8
    for (int e = 0; e < E_DIM; ++e) s += pooled[b * E_DIM + e] * Wc[c * E_DIM + e];
    out[b * 8 + c] = s;
}

// ---------------------------------------------------------------------------
extern "C" void kernel_launch(void* const* d_in, const int* in_sizes, int n_in,
                              void* d_out, int out_size, void* d_ws, size_t ws_size,
                              hipStream_t stream) {
    const int*   tokens = (const int*)d_in[0];
    const float* emb    = (const float*)d_in[1];
    const float* ln1_g  = (const float*)d_in[2];
    const float* ln1_b  = (const float*)d_in[3];
    const float* ln2_g  = (const float*)d_in[4];
    const float* ln2_b  = (const float*)d_in[5];
    const float* phi    = (const float*)d_in[6];   // (L, 8)
    const float* Wcomb  = (const float*)d_in[7];   // (L, E, E)
    const float* psi    = (const float*)d_in[8];   // (L, E)
    const float* W1     = (const float*)d_in[9];   // (L, F, E)
    const float* W2     = (const float*)d_in[10];  // (L, E, F)
    const float* Wc     = (const float*)d_in[11];  // (C, E)
    const float* bc     = (const float*)d_in[12];  // (C,)

    char* ws = (char*)d_ws;
    float*     x        = (float*)ws;                              // 16 MB
    _Float16*  q        = (_Float16*)(ws + ((size_t)16 << 20));    // 8 MB
    _Float16*  qT       = (_Float16*)(ws + ((size_t)24 << 20));    // 8 MB
    _Float16*  attn_out = (_Float16*)(ws + ((size_t)32 << 20));    // 8 MB
    _Float16*  qf       = (_Float16*)(ws + ((size_t)16 << 20));    // alias q
    _Float16*  gbuf     = (_Float16*)(ws + ((size_t)24 << 20));    // alias qT+attn, 32 MB
    float*     pooled   = (float*)(ws + ((size_t)56 << 20));       // 16 KB
    _Float16*  wcomb_h  = (_Float16*)(ws + ((size_t)57 << 20));    // 128 KB
    _Float16*  w1_h     = (_Float16*)(ws + ((size_t)58 << 20));    // 512 KB
    _Float16*  w2_h     = (_Float16*)(ws + ((size_t)59 << 20));    // 512 KB

    // One-shot weight conversion (f32 -> f16), all layers
    cvt_f16_kernel<<<256, 256, 0, stream>>>(Wcomb, wcomb_h, L_NUM * E_DIM * E_DIM);
    cvt_f16_kernel<<<512, 256, 0, stream>>>(W1, w1_h, L_NUM * F_DIM * E_DIM);
    cvt_f16_kernel<<<512, 256, 0, stream>>>(W2, w2_h, L_NUM * E_DIM * F_DIM);

    embed_kernel<<<BS, 128, 0, stream>>>(tokens, emb, x);

    for (int l = 0; l < L_NUM; ++l) {
        qgen_attn_kernel<<<BS / 8, 256, 0, stream>>>(
            x, ln1_g + l * E_DIM, ln1_b + l * E_DIM, phi + l * 8, q, qT);

        attn_kernel<<<(32 * (S_LEN / 16)) / 8, 256, 0, stream>>>(q, qT, attn_out);

        // x = x + attn_out @ Wcomb[l]^T        (M=BS, N=128, K=128)
        gemm_kernel<E_DIM, E_DIM, false, true, false>
            <<<(BS / 16) * (E_DIM / 64) / 8, 256, 0, stream>>>(
                attn_out, wcomb_h + (size_t)l * E_DIM * E_DIM, x, x, nullptr);

        qgen_ffn_kernel<<<BS / 8, 256, 0, stream>>>(
            x, ln2_g + l * E_DIM, ln2_b + l * E_DIM, psi + l * E_DIM, qf);

        // g = relu(qf @ W1[l]^T)               (M=BS, N=512, K=128) -> f16
        gemm_kernel<F_DIM, E_DIM, true, false, true>
            <<<(BS / 16) * (F_DIM / 64) / 8, 256, 0, stream>>>(
                qf, w1_h + (size_t)l * F_DIM * E_DIM, nullptr, nullptr, gbuf);

        // x = x + g @ W2[l]^T                  (M=BS, N=128, K=512)
        gemm_kernel<E_DIM, F_DIM, false, true, false>
            <<<(BS / 16) * (E_DIM / 64) / 8, 256, 0, stream>>>(
                gbuf, w2_h + (size_t)l * E_DIM * F_DIM, x, x, nullptr);
    }

    pool_kernel<<<32, 128, 0, stream>>>(x, pooled);
    cls_kernel<<<1, 256, 0, stream>>>(pooled, Wc, bc, (float*)d_out);
}
